// RecursiveCompressionLayer_10196252360938
// MI455X (gfx1250) — matrix operand order repaired and checked by reading, not compiled, not run on hardware
//
#include <hip/hip_runtime.h>
#include <math.h>

typedef __attribute__((ext_vector_type(16))) _Float16 v16h;
typedef __attribute__((ext_vector_type(8)))  _Float16 v8h;
typedef __attribute__((ext_vector_type(8)))  float    v8f;
typedef __attribute__((ext_vector_type(4)))  float    v4f;

#define NB   2
#define NS   2048
#define ND   1024
#define NDC  256
#define NDH  512
#define NBS  4096
#define HBINS 65536
#define SBLD 40   // LDS stride (halves) for K-major B tile: 80B, 16B-aligned

// ---------------- WMMA fragment helpers (ISA 7.12.2 layouts, wave32) ----------------

__device__ inline v16h make_v16(v8h a, v8h b) {
    return __builtin_shufflevector(a, b, 0, 1, 2, 3, 4, 5, 6, 7,
                                   8, 9, 10, 11, 12, 13, 14, 15);
}

// A: 16x32 f16. lanes 0-15: M=lane, K in {0..7,16..23}; lanes 16-31: M=lane-16,
// K in {8..15,24..31}. Per lane: two contiguous 8-half runs -> two b128 global loads.
__device__ inline v16h load_afrag_global(const _Float16* __restrict__ A,
                                         size_t rowOff, int kk, int lane) {
    const int kh = (lane >> 4) * 8;
    const _Float16* p = A + rowOff + kk + kh;
    v8h a0 = *(const v8h*)p;
    v8h a1 = *(const v8h*)(p + 16);
    return make_v16(a0, a1);
}

// B: 32x16 f16 (KxN). lanes 0-15 hold K in {0..7,16..23} of col n, lanes 16-31 K in {8..15,24..31}.
// LDS tile is K-major: sB[n*SBLD + k] -> per-lane fragment is two 8-half runs, 16 halves apart.
__device__ inline v16h load_bfrag_lds(const _Float16* sB, int n, int lane) {
    const _Float16* p = sB + n * SBLD + (lane >> 4) * 8;
    v8h b0 = *(const v8h*)p;
    v8h b1 = *(const v8h*)(p + 16);
    return make_v16(b0, b1);
}

__device__ inline float gelu_exact(float x) {
    return 0.5f * x * (1.0f + erff(x * 0.70710678118654752f));
}
__device__ inline float sigmoidf_(float x) { return 1.0f / (1.0f + __expf(-x)); }

// ---------------- B-tile fill (32 x 64, stored K-major in LDS) ----------------
// BMODE: 0 = f32 row-major [K,N] (weights); 1 = f16 row-major [K,N];
//        2 = f16 transposed source (B[k][n] = src[n*K + k]).
template <int BMODE>
__device__ inline void fill_btile(const void* __restrict__ Bsrc, _Float16* sB,
                                  int N, int K, int nBase, int kk, int tid) {
    if (BMODE == 0) {
        #pragma unroll
        for (int ii = 0; ii < 4; ++ii) {
            const int i  = tid + ii * 128;
            const int r  = i >> 4;          // k: 0..31
            const int c4 = (i & 15) * 4;    // n: 0..60
            v4f w = *(const v4f*)((const float*)Bsrc +
                                  (size_t)(kk + r) * N + nBase + c4);
            #pragma unroll
            for (int j = 0; j < 4; ++j) sB[(c4 + j) * SBLD + r] = (_Float16)w[j];
        }
    } else if (BMODE == 1) {
        #pragma unroll
        for (int ii = 0; ii < 2; ++ii) {
            const int i  = tid + ii * 128;
            const int r  = i >> 3;          // k: 0..31
            const int c8 = (i & 7) * 8;     // n: 0..56
            v8h w = *(const v8h*)((const _Float16*)Bsrc +
                                  (size_t)(kk + r) * N + nBase + c8);
            #pragma unroll
            for (int j = 0; j < 8; ++j) sB[(c8 + j) * SBLD + r] = w[j];
        }
    } else {
        #pragma unroll
        for (int ii = 0; ii < 2; ++ii) {
            const int i  = tid + ii * 128;
            const int c  = i >> 2;          // n: 0..63
            const int r8 = (i & 3) * 8;     // k: 0,8,16,24
            v8h w = *(const v8h*)((const _Float16*)Bsrc +
                                  (size_t)(nBase + c) * K + kk + r8);
            *(v8h*)(sB + c * SBLD + r8) = w;
        }
    }
}

// ---------------- shared 128x64-tile GEMM body (double-buffered LDS) ----------------
// 4 waves; each wave owns 32 rows (2 A fragments) x 64 cols (4 B sub-tiles):
// per K-chunk: 8 WMMAs against 8 ds_load_b128 + 4 global_load_b128.
template <int BMODE>
__device__ inline void gemm_tile_body(
    const _Float16* __restrict__ A, const void* __restrict__ Bsrc,
    _Float16* sB0, _Float16* sB1, int N, int K, int mBase, int nBase,
    int wave, int lane, int tid, v8f acc[8])
{
    const size_t row0 = (size_t)(mBase + wave * 32 + (lane & 15)) * K;
    const size_t row1 = row0 + (size_t)16 * K;
    fill_btile<BMODE>(Bsrc, sB0, N, K, nBase, 0, tid);
    __syncthreads();
    for (int kk = 0; kk < K; kk += 32) {
        _Float16* cur = ((kk >> 5) & 1) ? sB1 : sB0;
        _Float16* nxt = ((kk >> 5) & 1) ? sB0 : sB1;
        if (kk + 32 < K) fill_btile<BMODE>(Bsrc, nxt, N, K, nBase, kk + 32, tid);
        const v16h a0 = load_afrag_global(A, row0, kk, lane);
        const v16h a1 = load_afrag_global(A, row1, kk, lane);
        __builtin_prefetch((const void*)(A + row0 + kk + 64), 0, 0);
        #pragma unroll
        for (int nt = 0; nt < 4; ++nt) {
            const v16h bf = load_bfrag_lds(cur, nt * 16 + (lane & 15), lane);
            acc[nt] = __builtin_amdgcn_wmma_f32_16x16x32_f16(
                false, a0, false, bf, (short)0, acc[nt], false, false);
            acc[4 + nt] = __builtin_amdgcn_wmma_f32_16x16x32_f16(
                false, a1, false, bf, (short)0, acc[4 + nt], false, false);
        }
        __syncthreads();
    }
}

// ---------------- Generic GEMM: C = act(A[MxK] * B[KxN] + bias) ----------------
// ACT: 0=none, 1=gelu, 2=sigmoid.
template <int ACT, int BMODE, bool OUT_F16>
__global__ __launch_bounds__(128) void gemm_wmma_kernel(
    const _Float16* __restrict__ A, const void* __restrict__ Bsrc,
    const float* __restrict__ bias, void* __restrict__ Cout, int N, int K)
{
    __shared__ _Float16 sB0[64 * SBLD];
    __shared__ _Float16 sB1[64 * SBLD];
    const int tid = threadIdx.x, wave = tid >> 5, lane = tid & 31;
    const int mBase = blockIdx.y * 128, nBase = blockIdx.x * 64;

    v8f acc[8] = {};
    gemm_tile_body<BMODE>(A, Bsrc, sB0, sB1, N, K, mBase, nBase, wave, lane, tid, acc);

    const int nLane = lane & 15;
    const int mOff  = (lane >> 4) * 8;
    #pragma unroll
    for (int a = 0; a < 2; ++a) {
        #pragma unroll
        for (int nt = 0; nt < 4; ++nt) {
            const int gn = nBase + nt * 16 + nLane;
            const float bv = bias ? bias[gn] : 0.0f;
            #pragma unroll
            for (int i = 0; i < 8; ++i) {
                const int gm = mBase + wave * 32 + a * 16 + mOff + i;
                float v = acc[a * 4 + nt][i] + bv;
                if (ACT == 1) v = gelu_exact(v);
                else if (ACT == 2) v = sigmoidf_(v);
                if (OUT_F16) ((_Float16*)Cout)[(size_t)gm * N + gn] = (_Float16)v;
                else         ((float*)Cout)[(size_t)gm * N + gn] = v;
            }
        }
    }
}

// ---------------- pair = te_n * te_n^T, fused histogram + diag-block sim store ----------------
__global__ __launch_bounds__(128) void pair_hist_kernel(
    const _Float16* __restrict__ te, float* __restrict__ simbuf,
    unsigned int* __restrict__ hist)
{
    __shared__ _Float16 sB0[64 * SBLD];
    __shared__ _Float16 sB1[64 * SBLD];
    const int tid = threadIdx.x, wave = tid >> 5, lane = tid & 31;
    const int mBase = blockIdx.y * 128, nBase = blockIdx.x * 64;
    const bool diag = (mBase >> 11) == (nBase >> 11);

    v8f acc[8] = {};
    gemm_tile_body<2>(te, te, sB0, sB1, NBS, ND, mBase, nBase, wave, lane, tid, acc);

    const int nLane = lane & 15;
    const int mOff  = (lane >> 4) * 8;
    #pragma unroll
    for (int a = 0; a < 2; ++a) {
        #pragma unroll
        for (int nt = 0; nt < 4; ++nt) {
            const int gn = nBase + nt * 16 + nLane;
            #pragma unroll
            for (int i = 0; i < 8; ++i) {
                const int gm = mBase + wave * 32 + a * 16 + mOff + i;
                const float v = acc[a * 4 + nt][i];
                int bin = (int)((v + 1.0f) * (0.5f * (float)HBINS));
                bin = bin < 0 ? 0 : (bin > HBINS - 1 ? HBINS - 1 : bin);
                atomicAdd(&hist[bin], 1u);
                if (diag) {
                    const int b = gm >> 11;
                    simbuf[((size_t)b * NS + (gm & 2047)) * NS + (gn & 2047)] = v;
                }
            }
        }
    }
}

// ---------------- blended attention scores, in place over simbuf ----------------
// sim[b,s,t] <- 0.7*(q.k^T)/16 + 0.3*sim[b,s,t]*exp(-|s-t|/409.6)
__global__ __launch_bounds__(128) void scores_kernel(
    const _Float16* __restrict__ q, const _Float16* __restrict__ k,
    float* __restrict__ simbuf)
{
    __shared__ _Float16 sB0[64 * SBLD];
    __shared__ _Float16 sB1[64 * SBLD];
    const int tid = threadIdx.x, wave = tid >> 5, lane = tid & 31;
    const int b = blockIdx.z;
    const int mBase = blockIdx.y * 128, nBase = blockIdx.x * 64;
    const _Float16* qb = q + (size_t)b * NS * NDC;
    const _Float16* kb = k + (size_t)b * NS * NDC;
    float* sb = simbuf + (size_t)b * NS * NS;

    v8f acc[8] = {};
    gemm_tile_body<2>(qb, kb, sB0, sB1, NS, NDC, mBase, nBase, wave, lane, tid, acc);

    const int nLane = lane & 15;
    const int mOff  = (lane >> 4) * 8;
    #pragma unroll
    for (int a = 0; a < 2; ++a) {
        #pragma unroll
        for (int nt = 0; nt < 4; ++nt) {
            const int gn = nBase + nt * 16 + nLane;
            #pragma unroll
            for (int i = 0; i < 8; ++i) {
                const int gm = mBase + wave * 32 + a * 16 + mOff + i;
                const size_t idx = (size_t)gm * NS + gn;
                const float pen = __expf(-fabsf((float)(gm - gn)) * (1.0f / 409.6f));
                sb[idx] = 0.7f * (acc[a * 4 + nt][i] * 0.0625f) + 0.3f * sb[idx] * pen;
            }
        }
    }
}

// ---------------- per-token normalize + f16 copies ----------------
__global__ __launch_bounds__(256) void norm_tok_kernel(
    const float* __restrict__ tokens, _Float16* __restrict__ te16,
    _Float16* __restrict__ tok16)
{
    const size_t t = blockIdx.x;
    const float* x = tokens + t * ND;
    __shared__ float red[256];
    const int tid = threadIdx.x;
    float v[4], ss = 0.0f;
    #pragma unroll
    for (int j = 0; j < 4; ++j) { v[j] = x[tid + j * 256]; ss += v[j] * v[j]; }
    red[tid] = ss; __syncthreads();
    for (int off = 128; off > 0; off >>= 1) {
        if (tid < off) red[tid] += red[tid + off];
        __syncthreads();
    }
    const float inv = 1.0f / (sqrtf(red[0]) + 1e-12f);
    #pragma unroll
    for (int j = 0; j < 4; ++j) {
        te16[t * ND + tid + j * 256]  = (_Float16)(v[j] * inv);
        tok16[t * ND + tid + j * 256] = (_Float16)v[j];
    }
}

// ---------------- ctx[b,d] = mean_s tokens[b,s,d] ----------------
__global__ __launch_bounds__(256) void ctx_kernel(
    const float* __restrict__ tokens, float* __restrict__ ctx)
{
    const int idx = blockIdx.x * 256 + threadIdx.x;  // over B*D = 2048
    const int b = idx >> 10, d = idx & 1023;
    const float* p = tokens + (size_t)b * NS * ND + d;
    float s = 0.0f;
    for (int i = 0; i < NS; ++i) s += p[(size_t)i * ND];
    ctx[idx] = s * (1.0f / (float)NS);
}

// ---------------- quantile from histogram (one block) ----------------
__global__ __launch_bounds__(256) void quantile_kernel(
    const unsigned int* __restrict__ hist, float* __restrict__ scal)
{
    __shared__ unsigned long long csum[256];
    const int tid = threadIdx.x;
    unsigned long long s = 0;
    for (int i = 0; i < 256; ++i) s += hist[tid * 256 + i];
    csum[tid] = s; __syncthreads();
    if (tid == 0) {
        const unsigned long long N = (unsigned long long)NBS * NBS;
        const double pos = 0.95 * (double)(N - 1);
        const unsigned long long rank = (unsigned long long)pos;
        const double frac = pos - (double)rank;
        unsigned long long cum = 0; int chunk = 0;
        for (; chunk < 256; ++chunk) {
            if (cum + csum[chunk] > rank) break;
            cum += csum[chunk];
        }
        float val = 1.0f;
        for (int i = 0; i < 256; ++i) {
            const unsigned int h = hist[chunk * 256 + i];
            if (cum + h > rank) {
                const double within = (double)(rank - cum) + frac;
                const double f = (h > 0) ? within / (double)h : 0.5;
                const int bin = chunk * 256 + i;
                val = -1.0f + ((float)bin + (float)f) * (2.0f / (float)HBINS);
                break;
            }
            cum += h;
        }
        scal[0] = val;
    }
}

// ---------------- thr[b] = clip(0.75 + 0.25*sim_thr + sigmoid(ctx.w_mod+b)*0.1) ----------------
__global__ void thr_kernel(const float* __restrict__ ctx, const float* __restrict__ w_mod,
                           const float* __restrict__ b_mod, const float* __restrict__ scal,
                           float* __restrict__ thr)
{
    const int b = blockIdx.x;
    const int lane = threadIdx.x;
    float s = 0.0f;
    for (int i = lane; i < ND; i += 32) s += ctx[b * ND + i] * w_mod[i];
    for (int off = 16; off > 0; off >>= 1) s += __shfl_down(s, off);
    if (lane == 0) {
        const float tm = sigmoidf_(s + b_mod[0]) * 0.1f;
        float t = 0.75f + 0.25f * scal[0] + tm;
        thr[b] = fminf(fmaxf(t, 0.75f), 0.98f);
    }
}

// ---------------- row softmax: scores f32 -> attn f16 ----------------
__global__ __launch_bounds__(256) void softmax_kernel(
    const float* __restrict__ scores, _Float16* __restrict__ attn)
{
    const size_t row = blockIdx.x;
    const float* s = scores + row * NS;
    _Float16* a = attn + row * NS;
    __shared__ float red[256];
    const int tid = threadIdx.x;
    float mx = -1e30f;
    float v[8];
    #pragma unroll
    for (int j = 0; j < 8; ++j) { v[j] = s[tid + j * 256]; mx = fmaxf(mx, v[j]); }
    red[tid] = mx; __syncthreads();
    for (int off = 128; off > 0; off >>= 1) {
        if (tid < off) red[tid] = fmaxf(red[tid], red[tid + off]);
        __syncthreads();
    }
    mx = red[0]; __syncthreads();
    float sum = 0.0f;
    #pragma unroll
    for (int j = 0; j < 8; ++j) { v[j] = __expf(v[j] - mx); sum += v[j]; }
    red[tid] = sum; __syncthreads();
    for (int off = 128; off > 0; off >>= 1) {
        if (tid < off) red[tid] += red[tid + off];
        __syncthreads();
    }
    const float inv = 1.0f / red[0];
    #pragma unroll
    for (int j = 0; j < 8; ++j) a[tid + j * 256] = (_Float16)(v[j] * inv);
}

// ---------------- LayerNorm + residual: recon -> d_out, res16 = tokens - recon ----------------
__global__ __launch_bounds__(256) void ln_res_kernel(
    const float* __restrict__ pre, const float* __restrict__ tokens,
    const float* __restrict__ g, const float* __restrict__ beta,
    float* __restrict__ recon, _Float16* __restrict__ res16)
{
    const size_t t = blockIdx.x;
    const float* x = pre + t * ND;
    const float* tok = tokens + t * ND;
    __shared__ float red[256];
    const int tid = threadIdx.x;
    float v[4], s = 0.0f;
    #pragma unroll
    for (int j = 0; j < 4; ++j) { v[j] = x[tid + j * 256]; s += v[j]; }
    red[tid] = s; __syncthreads();
    for (int off = 128; off > 0; off >>= 1) {
        if (tid < off) red[tid] += red[tid + off];
        __syncthreads();
    }
    const float mean = red[0] * (1.0f / (float)ND);
    __syncthreads();
    s = 0.0f;
    #pragma unroll
    for (int j = 0; j < 4; ++j) { const float d = v[j] - mean; s += d * d; }
    red[tid] = s; __syncthreads();
    for (int off = 128; off > 0; off >>= 1) {
        if (tid < off) red[tid] += red[tid + off];
        __syncthreads();
    }
    const float inv = rsqrtf(red[0] * (1.0f / (float)ND) + 1e-5f);
    #pragma unroll
    for (int j = 0; j < 4; ++j) {
        const int d = tid + j * 256;
        const float r = (v[j] - mean) * inv * g[d] + beta[d];
        recon[t * ND + d] = r;
        res16[t * ND + d] = (_Float16)(tok[d] - r);
    }
}

// ---------------- imp[token] = sigmoid(tokens . w_imp + b_imp) ----------------
__global__ __launch_bounds__(256) void imp_kernel(
    const float* __restrict__ tokens, const float* __restrict__ w_imp,
    const float* __restrict__ b_imp, float* __restrict__ imp)
{
    const size_t t = blockIdx.x;
    const float* x = tokens + t * ND;
    __shared__ float red[256];
    const int tid = threadIdx.x;
    float s = 0.0f;
    #pragma unroll
    for (int j = 0; j < 4; ++j) s += x[tid + j * 256] * w_imp[tid + j * 256];
    red[tid] = s; __syncthreads();
    for (int off = 128; off > 0; off >>= 1) {
        if (tid < off) red[tid] += red[tid + off];
        __syncthreads();
    }
    if (tid == 0) imp[t] = sigmoidf_(red[0] + b_imp[0]);
}

// ---------------- out = recon + thr[b]*gate*res_c*imp ----------------
__global__ __launch_bounds__(256) void combine_kernel(
    float* __restrict__ out, const float* __restrict__ gate,
    const float* __restrict__ resc, const float* __restrict__ imp,
    const float* __restrict__ thr)
{
    const size_t i = (size_t)blockIdx.x * 256 + threadIdx.x;  // over BS*D
    const size_t tkn = i >> 10;
    const int b = (int)(tkn >> 11);
    out[i] = out[i] + thr[b] * gate[i] * resc[i] * imp[tkn];
}

// ================================================================================

extern "C" void kernel_launch(void* const* d_in, const int* in_sizes, int n_in,
                              void* d_out, int out_size, void* d_ws, size_t ws_size,
                              hipStream_t stream) {
    const float* tokens = (const float*)d_in[0];
    const float* w_mod  = (const float*)d_in[1];
    const float* b_mod  = (const float*)d_in[2];
    const float* w_comp = (const float*)d_in[3];
    const float* b_comp = (const float*)d_in[4];
    const float* w_q    = (const float*)d_in[5];
    const float* b_q    = (const float*)d_in[6];
    const float* w_k    = (const float*)d_in[7];
    const float* b_k    = (const float*)d_in[8];
    const float* w_v    = (const float*)d_in[9];
    const float* b_v    = (const float*)d_in[10];
    const float* w_m1   = (const float*)d_in[11];
    const float* b_m1   = (const float*)d_in[12];
    const float* w_m2   = (const float*)d_in[13];
    const float* b_m2   = (const float*)d_in[14];
    const float* ln_g   = (const float*)d_in[15];
    const float* ln_b   = (const float*)d_in[16];
    const float* w_rc1  = (const float*)d_in[17];
    const float* b_rc1  = (const float*)d_in[18];
    const float* w_rc2  = (const float*)d_in[19];
    const float* b_rc2  = (const float*)d_in[20];
    const float* w_rg1  = (const float*)d_in[21];
    const float* b_rg1  = (const float*)d_in[22];
    const float* w_rg2  = (const float*)d_in[23];
    const float* b_rg2  = (const float*)d_in[24];
    const float* w_imp  = (const float*)d_in[25];
    const float* b_imp  = (const float*)d_in[26];
    float* out = (float*)d_out;

    char* ws = (char*)d_ws;
    size_t off = 0;
    auto take = [&](size_t bytes) { char* p = ws + off; off += (bytes + 255) & ~(size_t)255; return p; };
    _Float16* te16   = (_Float16*)take((size_t)NBS * ND * 2);
    _Float16* tok16  = (_Float16*)take((size_t)NBS * ND * 2);
    float*    simbuf = (float*)   take((size_t)NB * NS * NS * 4);
    _Float16* attn16 = (_Float16*)take((size_t)NB * NS * NS * 2);
    unsigned int* hist = (unsigned int*)take((size_t)HBINS * 4);
    float*    ctx    = (float*)   take((size_t)NB * ND * 4);
    float*    scal   = (float*)   take(256);                 // [0]=sim_thr, [8..9]=thr
    _Float16* comp16 = (_Float16*)take((size_t)NBS * NDC * 2);
    _Float16* q16    = (_Float16*)take((size_t)NBS * NDC * 2);
    _Float16* k16    = (_Float16*)take((size_t)NBS * NDC * 2);
    _Float16* v16    = (_Float16*)take((size_t)NBS * NDC * 2);
    _Float16* ctxv16 = (_Float16*)take((size_t)NBS * NDC * 2);
    _Float16* h16    = (_Float16*)take((size_t)NBS * NDH * 2);
    float*    pre    = (float*)   take((size_t)NBS * ND * 4);
    _Float16* res16  = (_Float16*)take((size_t)NBS * ND * 2);
    _Float16* g1_16  = (_Float16*)take((size_t)NBS * NDC * 2);
    float*    gate   = (float*)   take((size_t)NBS * ND * 4);
    _Float16* r1_16  = (_Float16*)take((size_t)NBS * NDC * 2);
    float*    resc   = (float*)   take((size_t)NBS * ND * 4);
    float*    imp    = (float*)   take((size_t)NBS * 4);
    float*    thr    = scal + 8;

    hipMemsetAsync(hist, 0, (size_t)HBINS * 4, stream);

    // 1. normalize + f16 copies, batch context means
    norm_tok_kernel<<<NBS, 256, 0, stream>>>(tokens, te16, tok16);
    ctx_kernel<<<(NB * ND) / 256, 256, 0, stream>>>(tokens, ctx);

    // 2. global pairwise cosine sims: histogram + diagonal-block sim store
    pair_hist_kernel<<<dim3(NBS / 64, NBS / 128), 128, 0, stream>>>(te16, simbuf, hist);
    quantile_kernel<<<1, 256, 0, stream>>>(hist, scal);
    thr_kernel<<<NB, 32, 0, stream>>>(ctx, w_mod, b_mod, scal, thr);

    // 3. compression + QKV
    gemm_wmma_kernel<0, 0, true><<<dim3(NDC / 64, NBS / 128), 128, 0, stream>>>(
        tok16, w_comp, b_comp, comp16, NDC, ND);
    gemm_wmma_kernel<0, 0, true><<<dim3(NDC / 64, NBS / 128), 128, 0, stream>>>(
        comp16, w_q, b_q, q16, NDC, NDC);
    gemm_wmma_kernel<0, 0, true><<<dim3(NDC / 64, NBS / 128), 128, 0, stream>>>(
        comp16, w_k, b_k, k16, NDC, NDC);
    gemm_wmma_kernel<0, 0, true><<<dim3(NDC / 64, NBS / 128), 128, 0, stream>>>(
        comp16, w_v, b_v, v16, NDC, NDC);

    // 4. attention: blended scores (in place over simbuf), softmax, attn @ v
    scores_kernel<<<dim3(NS / 64, NS / 128, NB), 128, 0, stream>>>(q16, k16, simbuf);
    softmax_kernel<<<NB * NS, 256, 0, stream>>>(simbuf, attn16);
    for (int b = 0; b < NB; ++b) {
        gemm_wmma_kernel<0, 1, true><<<dim3(NDC / 64, NS / 128), 128, 0, stream>>>(
            attn16 + (size_t)b * NS * NS, v16 + (size_t)b * NS * NDC, nullptr,
            ctxv16 + (size_t)b * NS * NDC, NDC, NS);
    }

    // 5. decompression MLP + LayerNorm, residual
    gemm_wmma_kernel<1, 0, true><<<dim3(NDH / 64, NBS / 128), 128, 0, stream>>>(
        ctxv16, w_m1, b_m1, h16, NDH, NDC);
    gemm_wmma_kernel<0, 0, false><<<dim3(ND / 64, NBS / 128), 128, 0, stream>>>(
        h16, w_m2, b_m2, pre, ND, NDH);
    ln_res_kernel<<<NBS, 256, 0, stream>>>(pre, tokens, ln_g, ln_b, out, res16);

    // 6. gating + residual-compression paths
    gemm_wmma_kernel<1, 0, true><<<dim3(NDC / 64, NBS / 128), 128, 0, stream>>>(
        tok16, w_rg1, b_rg1, g1_16, NDC, ND);
    gemm_wmma_kernel<2, 0, false><<<dim3(ND / 64, NBS / 128), 128, 0, stream>>>(
        g1_16, w_rg2, b_rg2, gate, ND, NDC);
    gemm_wmma_kernel<1, 0, true><<<dim3(NDC / 64, NBS / 128), 128, 0, stream>>>(
        res16, w_rc1, b_rc1, r1_16, NDC, ND);
    gemm_wmma_kernel<0, 0, false><<<dim3(ND / 64, NBS / 128), 128, 0, stream>>>(
        r1_16, w_rc2, b_rc2, resc, ND, NDC);

    // 7. importance + final combine
    imp_kernel<<<NBS, 256, 0, stream>>>(tokens, w_imp, b_imp, imp);
    combine_kernel<<<(NBS * ND) / 256, 256, 0, stream>>>(out, gate, resc, imp, thr);
}
